// RoPEAttention_61100204753731
// MI455X (gfx1250) — matrix operand order repaired and checked by reading, not compiled, not run on hardware
//
#include <hip/hip_runtime.h>

typedef __attribute__((ext_vector_type(16))) _Float16 v16h;
typedef __attribute__((ext_vector_type(8)))  float    v8f;
typedef unsigned int u32x4 __attribute__((ext_vector_type(4)));
typedef int          i32x4 __attribute__((ext_vector_type(4)));
typedef int          i32x8 __attribute__((ext_vector_type(8)));

#define BB 2
#define TT 2048
#define DD 1024
#define HH 16
#define HD 64
#define BT (BB * TT)

union FragH { uint4 u[2]; v16h h; };

__device__ __forceinline__ v8f wmma16(v16h a, v16h b, v8f c) {
  // D = A(16x32 f16) x B(32x16 f16) + C(16x16 f32)
  return __builtin_amdgcn_wmma_f32_16x16x32_f16(false, a, false, b, (short)0, c,
                                                false, false);
}

// TDM: DMA a 64-row x 32-col fp16 tile (row stride 1024 elems) into LDS.
// D# layout per CDNA5 ISA; 6-arg builtin form.
__device__ __forceinline__ void tdm_load_slab(const _Float16* gsrc,
                                              unsigned lds_addr) {
  const unsigned long long ga = (unsigned long long)(size_t)gsrc;
  u32x4 g0;
  g0[0] = 1u;                                 // count=1 valid user descriptor
  g0[1] = lds_addr;                           // lds_addr (bytes)
  g0[2] = (unsigned)ga;                       // global_addr[31:0]
  g0[3] = (unsigned)(ga >> 32) | (2u << 30);  // global_addr[56:32] | type=2
  i32x8 g1;
  g1[0] = 0x10000;             // data_size=1 -> 2-byte elements
  g1[1] = (int)(1024u << 16);  // tensor_dim0 = 1024 (bits 79:48)
  g1[2] = (int)(1024u << 16);  // tensor_dim1 = 1024 (bits 111:80)
  g1[3] = (int)(32u << 16);    // tile_dim0 = 32     (bits 127:112)
  g1[4] = 64;                  // tile_dim1 = 64, tile_dim2 = 0
  g1[5] = 1024;                // tensor_dim0_stride = 1024 (bits 207:160)
  g1[6] = 0;                   // stride hi / tensor_dim1_stride lo
  g1[7] = 0;
  i32x4 z4 = {0, 0, 0, 0};     // groups 2/3 unused (2-D tile)
  i32x8 z8 = {0, 0, 0, 0, 0, 0, 0, 0};
  __builtin_amdgcn_tensor_load_to_lds(g0, g1, z4, z4, z8, 0);
}

// ---------------------------------------------------------------- fp32 -> fp16
__global__ void cvt_kernel(const float* __restrict__ s, _Float16* __restrict__ d,
                           int n) {
  int i = blockIdx.x * blockDim.x + threadIdx.x;
  if (i < n) d[i] = (_Float16)s[i];
}

// ------------------------------------------------- QKV projection + RoPE fused
// Grid: 3 (mat) x 16 (head/N-slab) x 32 (M-block of 128 rows); 256 thr = 8 waves.
// Each wave: 16x64 output tile; W slabs DMA'd by the TDM into double-buffered
// LDS while the waves run WMMAs on the current slab.
__global__ __launch_bounds__(256)
void qkv_rope_kernel(const _Float16* __restrict__ xh,
                     const _Float16* __restrict__ Wqh,
                     const _Float16* __restrict__ Wkh,
                     const _Float16* __restrict__ Wvh,
                     const float* __restrict__ bq, const float* __restrict__ bk,
                     const float* __restrict__ bv,
                     _Float16* __restrict__ Qh, _Float16* __restrict__ Kh,
                     _Float16* __restrict__ Vt) {
  __shared__ alignas(16) _Float16 Wslab[2][64 * 32];

  const int bid  = blockIdx.x;
  const int mat  = bid / (16 * 32);      // 0=Q 1=K 2=V
  const int rem  = bid % (16 * 32);
  const int head = rem >> 5;             // 64-col N-slab == one head
  const int mblk = rem & 31;

  const _Float16* Wh = (mat == 0) ? Wqh : (mat == 1) ? Wkh : Wvh;

  const int lane = threadIdx.x & 31;
  const int wave = threadIdx.x >> 5;
  const int g = lane >> 4;               // half-wave group
  const int c = lane & 15;               // column / row-in-tile index
  const int row0 = mblk * 128 + wave * 16;
  const int n0 = head * 64;

  const _Float16* Wbase = Wh + (size_t)n0 * DD;  // rows n0..n0+63

  if (threadIdx.x < 32) {  // wave 0 owns the TDM pipe
    tdm_load_slab(Wbase, (unsigned)(size_t)&Wslab[0][0]);
    __builtin_amdgcn_s_wait_tensorcnt(0);
  }
  __syncthreads();

  v8f acc[4] = {};

  for (int k0 = 0; k0 < DD; k0 += 32) {
    const int cur = (k0 >> 5) & 1;
    if (threadIdx.x < 32 && (k0 + 32) < DD)  // prefetch next slab via TDM
      tdm_load_slab(Wbase + (k0 + 32), (unsigned)(size_t)&Wslab[cur ^ 1][0]);

    FragH a;  // A 16x32: lane row c, contiguous k chunk at 16g
    const _Float16* xrow = xh + (size_t)(row0 + c) * DD + k0;
    a.u[0] = *(const uint4*)(xrow + g * 16);
    a.u[1] = *(const uint4*)(xrow + g * 16 + 8);

    const _Float16* wb = &Wslab[cur][0];
#pragma unroll
    for (int nt = 0; nt < 4; ++nt) {
      FragH bfr;  // B 32x16: lane col nt*16+c, contiguous k chunk at 16g
      const _Float16* wc = wb + (nt * 16 + c) * 32 + g * 16;
      bfr.u[0] = *(const uint4*)(wc);
      bfr.u[1] = *(const uint4*)(wc + 8);
      acc[nt] = wmma16(a.h, bfr.h, acc[nt]);
    }

    if (threadIdx.x < 32) __builtin_amdgcn_s_wait_tensorcnt(0);
    __syncthreads();
  }

  const int bb = row0 / TT;
  const int tbase = row0 % TT;

  if (mat < 2) {
    // RoPE: lane holds dims {c, c+16, c+32, c+48}; pairs (c,c+32) freq c and
    // (c+16,c+48) freq c+16 are lane-local in the D layout.
    const float* bias = (mat == 0) ? bq : bk;
    _Float16* Out = ((mat == 0) ? Qh : Kh) + (size_t)(bb * HH + head) * TT * HD;
    const float b0 = bias[n0 + c], b1 = bias[n0 + 16 + c];
    const float b2 = bias[n0 + 32 + c], b3 = bias[n0 + 48 + c];
    const float LN = 9.210340371976184f;  // ln(10000)
    const float invf0 = __expf(-((float)(2 * c) * (1.0f / 64.f)) * LN);
    const float invf1 = __expf(-((float)(2 * (c + 16)) * (1.0f / 64.f)) * LN);
#pragma unroll
    for (int r = 0; r < 8; ++r) {
      const int t = tbase + g * 8 + r;
      float s0, c0, s1, c1;
      __sincosf((float)t * invf0, &s0, &c0);
      __sincosf((float)t * invf1, &s1, &c1);
      const float x0 = acc[0][r] + b0, x1 = acc[1][r] + b1;
      const float x2 = acc[2][r] + b2, x3 = acc[3][r] + b3;
      const size_t base = (size_t)t * HD;
      Out[base + c]      = (_Float16)(x0 * c0 - x2 * s0);
      Out[base + 16 + c] = (_Float16)(x1 * c1 - x3 * s1);
      Out[base + 32 + c] = (_Float16)(x0 * s0 + x2 * c0);
      Out[base + 48 + c] = (_Float16)(x1 * s1 + x3 * c1);
    }
  } else {
    // V stored transposed [b][h][hd][T] so PV B-fragments are contiguous.
    _Float16* Out = Vt + (size_t)(bb * HH + head) * HD * TT;
    const float b0 = bv[n0 + c], b1 = bv[n0 + 16 + c];
    const float b2 = bv[n0 + 32 + c], b3 = bv[n0 + 48 + c];
#pragma unroll
    for (int r = 0; r < 8; ++r) {
      const int t = tbase + g * 8 + r;
      Out[(size_t)c * TT + t]        = (_Float16)(acc[0][r] + b0);
      Out[(size_t)(c + 16) * TT + t] = (_Float16)(acc[1][r] + b1);
      Out[(size_t)(c + 32) * TT + t] = (_Float16)(acc[2][r] + b2);
      Out[(size_t)(c + 48) * TT + t] = (_Float16)(acc[3][r] + b3);
    }
  }
}

// --------------------------------------------- flash attention, one wave/qtile
__global__ __launch_bounds__(32)
void flash_attn_kernel(const _Float16* __restrict__ Qh,
                       const _Float16* __restrict__ Kh,
                       const _Float16* __restrict__ Vt,
                       _Float16* __restrict__ Ah) {
  __shared__ alignas(16) float    Sbuf[16 * 32];
  __shared__ alignas(16) _Float16 Pbuf[16 * 32];
  __shared__ float RowTmp[16];

  const int bid  = blockIdx.x;
  const int qt   = bid & 127;
  const int head = (bid >> 7) & 15;
  const int bb   = bid >> 11;
  const int q0   = qt * 16;
  const int lane = threadIdx.x;
  const int g = lane >> 4, c = lane & 15;

  const _Float16* Qp = Qh + (size_t)(bb * HH + head) * TT * HD;
  const _Float16* Kp = Kh + (size_t)(bb * HH + head) * TT * HD;
  const _Float16* Vp = Vt + (size_t)(bb * HH + head) * HD * TT;

  // Q A-fragments for hd k-chunks [0,32) and [32,64)
  FragH qf0, qf1;
  const _Float16* qrow = Qp + (size_t)(q0 + c) * HD;
  qf0.u[0] = *(const uint4*)(qrow + g * 16);
  qf0.u[1] = *(const uint4*)(qrow + g * 16 + 8);
  qf1.u[0] = *(const uint4*)(qrow + 32 + g * 16);
  qf1.u[1] = *(const uint4*)(qrow + 32 + g * 16 + 8);

  v8f o[4] = {};
  float m_run = -1e30f, l_run = 0.f;
  const int tq = q0 + c;

  for (int kt0 = 0; kt0 < q0 + 16; kt0 += 32) {
    // ---- S = Q K^T (16 x 32 over this kv block, K-dim = hd = 64)
#pragma unroll
    for (int nt = 0; nt < 2; ++nt) {
      FragH kf0, kf1;  // B frags: col = kv row, k = hd contiguous chunk
      const _Float16* kcol = Kp + (size_t)(kt0 + nt * 16 + c) * HD;
      kf0.u[0] = *(const uint4*)(kcol + g * 16);
      kf0.u[1] = *(const uint4*)(kcol + g * 16 + 8);
      kf1.u[0] = *(const uint4*)(kcol + 32 + g * 16);
      kf1.u[1] = *(const uint4*)(kcol + 32 + g * 16 + 8);
      v8f s = {};
      s = wmma16(qf0.h, kf0.h, s);
      s = wmma16(qf1.h, kf1.h, s);
#pragma unroll
      for (int r = 0; r < 8; ++r)
        Sbuf[(g * 8 + r) * 32 + nt * 16 + c] = s[r];
    }
    __syncthreads();

    // ---- online softmax: lane owns row c, half g (cols 16g..16g+15)
    float sv[16];
    float mloc = -1e30f;
#pragma unroll
    for (int j = 0; j < 16; ++j) {
      const int n = g * 16 + j;
      float v = Sbuf[c * 32 + n] * 0.125f;          // 1/sqrt(64)
      v = fminf(fmaxf(v, -10000.f), 10000.f);       // clamp
      if (kt0 + n > tq) v = -10000.f;               // causal: -inf then clip
      sv[j] = v;
      mloc = fmaxf(mloc, v);
    }
    mloc = fmaxf(mloc, __shfl_xor(mloc, 16, 32));
    const float mnew  = fmaxf(m_run, mloc);
    const float alpha = __expf(m_run - mnew);
    float lsum = 0.f;
#pragma unroll
    for (int j = 0; j < 16; ++j) {
      const float pj = __expf(sv[j] - mnew);
      lsum += pj;
      Pbuf[c * 32 + g * 16 + j] = (_Float16)pj;
    }
    lsum += __shfl_xor(lsum, 16, 32);
    l_run = l_run * alpha + lsum;
    m_run = mnew;
    if (g == 0) RowTmp[c] = alpha;
    __syncthreads();

    // ---- rescale O and accumulate O += P * V
    float al[8];
#pragma unroll
    for (int r = 0; r < 8; ++r) al[r] = RowTmp[g * 8 + r];
#pragma unroll
    for (int a = 0; a < 4; ++a)
#pragma unroll
      for (int r = 0; r < 8; ++r) o[a][r] *= al[r];

    FragH pf;  // P as A 16x32 fragment from LDS
    pf.u[0] = *(const uint4*)&Pbuf[c * 32 + g * 16];
    pf.u[1] = *(const uint4*)&Pbuf[c * 32 + g * 16 + 8];
#pragma unroll
    for (int a = 0; a < 4; ++a) {
      FragH vf;  // V^T B frag: col = hd index a*16+c, k = kv contiguous
      const _Float16* vcol = Vp + (size_t)(a * 16 + c) * TT + kt0 + g * 16;
      vf.u[0] = *(const uint4*)(vcol);
      vf.u[1] = *(const uint4*)(vcol + 8);
      o[a] = wmma16(pf.h, vf.h, o[a]);
    }
    __syncthreads();
  }

  if (g == 0) RowTmp[c] = 1.0f / l_run;
  __syncthreads();
  float li[8];
#pragma unroll
  for (int r = 0; r < 8; ++r) li[r] = RowTmp[g * 8 + r];
#pragma unroll
  for (int r = 0; r < 8; ++r) {
    const int t = q0 + g * 8 + r;
    const size_t base = ((size_t)bb * TT + t) * DD + head * 64;
#pragma unroll
    for (int a = 0; a < 4; ++a)
      Ah[base + a * 16 + c] = (_Float16)(o[a][r] * li[r]);
  }
}

// ---------------------------------------------------------- output projection
__global__ __launch_bounds__(256)
void out_proj_kernel(const _Float16* __restrict__ Ah,
                     const _Float16* __restrict__ Woh,
                     const float* __restrict__ bo, float* __restrict__ out) {
  __shared__ alignas(16) _Float16 Wslab[2][64 * 32];
  const int bid = blockIdx.x;
  const int nslab = bid & 15;
  const int mblk = bid >> 4;
  const int lane = threadIdx.x & 31;
  const int wave = threadIdx.x >> 5;
  const int g = lane >> 4, c = lane & 15;
  const int row0 = mblk * 128 + wave * 16;
  const int n0 = nslab * 64;

  const _Float16* Wbase = Woh + (size_t)n0 * DD;

  if (threadIdx.x < 32) {
    tdm_load_slab(Wbase, (unsigned)(size_t)&Wslab[0][0]);
    __builtin_amdgcn_s_wait_tensorcnt(0);
  }
  __syncthreads();

  v8f acc[4] = {};

  for (int k0 = 0; k0 < DD; k0 += 32) {
    const int cur = (k0 >> 5) & 1;
    if (threadIdx.x < 32 && (k0 + 32) < DD)
      tdm_load_slab(Wbase + (k0 + 32), (unsigned)(size_t)&Wslab[cur ^ 1][0]);

    FragH a;
    const _Float16* arow = Ah + (size_t)(row0 + c) * DD + k0;
    a.u[0] = *(const uint4*)(arow + g * 16);
    a.u[1] = *(const uint4*)(arow + g * 16 + 8);

    const _Float16* wb = &Wslab[cur][0];
#pragma unroll
    for (int nt = 0; nt < 4; ++nt) {
      FragH bfr;
      const _Float16* wc = wb + (nt * 16 + c) * 32 + g * 16;
      bfr.u[0] = *(const uint4*)(wc);
      bfr.u[1] = *(const uint4*)(wc + 8);
      acc[nt] = wmma16(a.h, bfr.h, acc[nt]);
    }

    if (threadIdx.x < 32) __builtin_amdgcn_s_wait_tensorcnt(0);
    __syncthreads();
  }

#pragma unroll
  for (int r = 0; r < 8; ++r) {
    const int row = row0 + g * 8 + r;
#pragma unroll
    for (int a = 0; a < 4; ++a)
      out[(size_t)row * DD + n0 + a * 16 + c] = acc[a][r] + bo[n0 + a * 16 + c];
  }
}

// ----------------------------------------------------------------------------
extern "C" void kernel_launch(void* const* d_in, const int* in_sizes, int n_in,
                              void* d_out, int out_size, void* d_ws,
                              size_t ws_size, hipStream_t stream) {
  (void)in_sizes; (void)n_in; (void)out_size; (void)ws_size;
  const float* x  = (const float*)d_in[0];
  // d_in[1] = causal_mask (implicit in kernel)
  const float* Wq = (const float*)d_in[2];
  const float* bq = (const float*)d_in[3];
  const float* Wk = (const float*)d_in[4];
  const float* bk = (const float*)d_in[5];
  const float* Wv = (const float*)d_in[6];
  const float* bv = (const float*)d_in[7];
  const float* Wo = (const float*)d_in[8];
  const float* bo = (const float*)d_in[9];

  // workspace carve (~48 MB total)
  char* p = (char*)d_ws;
  const size_t szX = (size_t)BT * DD * sizeof(_Float16);  // 8 MB
  const size_t szW = (size_t)DD * DD * sizeof(_Float16);  // 2 MB
  _Float16* xh  = (_Float16*)p; p += szX;
  _Float16* Wqh = (_Float16*)p; p += szW;
  _Float16* Wkh = (_Float16*)p; p += szW;
  _Float16* Wvh = (_Float16*)p; p += szW;
  _Float16* Woh = (_Float16*)p; p += szW;
  _Float16* Qh  = (_Float16*)p; p += szX;
  _Float16* Kh  = (_Float16*)p; p += szX;
  _Float16* Vt  = (_Float16*)p; p += szX;
  _Float16* Ah  = (_Float16*)p; p += szX;

  const int nX = BT * DD, nW = DD * DD;
  cvt_kernel<<<(nX + 255) / 256, 256, 0, stream>>>(x, xh, nX);
  cvt_kernel<<<(nW + 255) / 256, 256, 0, stream>>>(Wq, Wqh, nW);
  cvt_kernel<<<(nW + 255) / 256, 256, 0, stream>>>(Wk, Wkh, nW);
  cvt_kernel<<<(nW + 255) / 256, 256, 0, stream>>>(Wv, Wvh, nW);
  cvt_kernel<<<(nW + 255) / 256, 256, 0, stream>>>(Wo, Woh, nW);

  qkv_rope_kernel<<<3 * 16 * 32, 256, 0, stream>>>(xh, Wqh, Wkh, Wvh, bq, bk,
                                                   bv, Qh, Kh, Vt);
  flash_attn_kernel<<<BB * HH * (TT / 16), 32, 0, stream>>>(Qh, Kh, Vt, Ah);
  out_proj_kernel<<<(BT / 128) * (DD / 64), 256, 0, stream>>>(Ah, Woh, bo,
                                                              (float*)d_out);
}
